// GBFormer_59270548685257
// MI455X (gfx1250) — matrix operand order repaired and checked by reading, not compiled, not run on hardware
//
#include <hip/hip_runtime.h>
#include <hip/hip_bf16.h>
#include <math.h>

typedef __bf16 bf16_t;
typedef bf16_t v16bf  __attribute__((ext_vector_type(16)));
typedef bf16_t bf16x8 __attribute__((ext_vector_type(8)));
typedef bf16_t bf16x4 __attribute__((ext_vector_type(4)));
typedef bf16_t bf16x2 __attribute__((ext_vector_type(2)));
typedef float  v8f    __attribute__((ext_vector_type(8)));

#define BB    4
#define GG    18432
#define DD    128
#define DEG   20
#define NBINS 6
#define NBIN  3072
#define DH    32
#define HEADS 2
#define INNER 64            // HEADS*DH
#define MFEAT 110           // int(32*ln(32))
#define MPAD  128
#define SORTN 32768
#define FFNH  512

#define CDIV(a,b) (((a)+(b)-1)/(b))

// ---------------- wave32 reductions ----------------
__device__ __forceinline__ float wsum(float v){
  #pragma unroll
  for(int o=16;o>0;o>>=1) v += __shfl_xor(v,o,32);
  return v;
}
__device__ __forceinline__ float wmaxr(float v){
  #pragma unroll
  for(int o=16;o>0;o>>=1) v = fmaxf(v,__shfl_xor(v,o,32));
  return v;
}
__device__ __forceinline__ unsigned fenc(float f){
  unsigned u=__float_as_uint(f); return (u>>31)? ~u : (u|0x80000000u);
}
__device__ __forceinline__ float fdec(unsigned e){
  unsigned u = (e & 0x80000000u) ? (e ^ 0x80000000u) : ~e;
  return __uint_as_float(u);
}

// =====================================================================
// Generic batched bf16 WMMA GEMM:  C[z] = epi( alpha * A[z] @ W[z] )
// slice offset: off(z) = (z/zsub)*S1 + (z%zsub)*S2 for each operand.
// Preconditions: M % 64 == 0, K % 32 == 0 (true for every GEMM here).
// Only N may be ragged (template NB).
// =====================================================================
struct GemmP {
  const float* A; int lda; long long aS1, aS2;
  const float* W; int ldw; long long wS1, wS2;
  float*       C; int ldc; long long cS1, cS2;
  const float* R; long long rS1, rS2;   // residual (same geometry as C), may be null
  const float* bias;                    // per output column, may be null
  int M,N,K,zsub,gelu;
  float alpha;
};

template<bool TA, bool TW, bool NB>
__global__ __launch_bounds__(128) void gemm_bf16_wmma(GemmP p){
  __shared__ bf16_t la[64*32];   // [m][k]
  __shared__ bf16_t lb[64*32];   // [n][k]  (W staged transposed)
  const int z = blockIdx.z;
  const long long zq = z / p.zsub, zr = z % p.zsub;
  const float* Ap = p.A + zq*p.aS1 + zr*p.aS2;
  const float* Wp = p.W + zq*p.wS1 + zr*p.wS2;
  const int m0 = blockIdx.x * 64, n0 = blockIdx.y * 64;
  const int t = threadIdx.x, wave = t>>5, lane = t&31;

  v8f acc[4];
  #pragma unroll
  for(int c=0;c<4;c++){
    #pragma unroll
    for(int r=0;r<8;r++) acc[c][r] = 0.f;
  }

  for(int k0=0;k0<p.K;k0+=32){
    // ---- stage A tile (64 m x 32 k): loads hoisted, then convert+store ----
    if(!TA){
      float4 va[4];
      #pragma unroll
      for(int i=0;i<4;i++){
        int r  = i*16 + (t>>3);
        int kq = (t&7)*4;
        va[i] = *(const float4*)(Ap + (long long)(m0+r)*p.lda + k0 + kq);
      }
      #pragma unroll
      for(int i=0;i<4;i++){
        int r  = i*16 + (t>>3);
        int kq = (t&7)*4;
        bf16x4 o; o[0]=(bf16_t)va[i].x; o[1]=(bf16_t)va[i].y;
                  o[2]=(bf16_t)va[i].z; o[3]=(bf16_t)va[i].w;
        *(bf16x4*)(la + r*32 + kq) = o;
      }
    } else {
      // A[m][k] = Ap[k*lda + m]: paired k-rows -> packed bf16x2 stores
      float4 a0[2], a1[2];
      #pragma unroll
      for(int i=0;i<2;i++){
        int kk = (t>>4)*2 + i*16;
        int mq = (t&15)*4;
        a0[i] = *(const float4*)(Ap + (long long)(k0+kk  )*p.lda + m0 + mq);
        a1[i] = *(const float4*)(Ap + (long long)(k0+kk+1)*p.lda + m0 + mq);
      }
      #pragma unroll
      for(int i=0;i<2;i++){
        int kk = (t>>4)*2 + i*16;
        int mq = (t&15)*4;
        const float* r0 = (const float*)&a0[i];
        const float* r1 = (const float*)&a1[i];
        #pragma unroll
        for(int j=0;j<4;j++){
          bf16x2 o; o[0]=(bf16_t)r0[j]; o[1]=(bf16_t)r1[j];
          *(bf16x2*)(la + (mq+j)*32 + kk) = o;
        }
      }
    }
    // ---- stage W tile into lb[n][k] ----
    if(!TW){
      // paired k-rows -> packed bf16x2 stores into transposed layout
      float4 w0[2], w1[2];
      #pragma unroll
      for(int i=0;i<2;i++){
        int kk = (t>>4)*2 + i*16;
        int nq = (t&15)*4;
        if(NB){
          const float* wr0 = Wp + (long long)(k0+kk  )*p.ldw;
          const float* wr1 = Wp + (long long)(k0+kk+1)*p.ldw;
          int n = n0 + nq;
          w0[i].x=(n+0<p.N)?wr0[n+0]:0.f; w0[i].y=(n+1<p.N)?wr0[n+1]:0.f;
          w0[i].z=(n+2<p.N)?wr0[n+2]:0.f; w0[i].w=(n+3<p.N)?wr0[n+3]:0.f;
          w1[i].x=(n+0<p.N)?wr1[n+0]:0.f; w1[i].y=(n+1<p.N)?wr1[n+1]:0.f;
          w1[i].z=(n+2<p.N)?wr1[n+2]:0.f; w1[i].w=(n+3<p.N)?wr1[n+3]:0.f;
        } else {
          w0[i] = *(const float4*)(Wp + (long long)(k0+kk  )*p.ldw + n0 + nq);
          w1[i] = *(const float4*)(Wp + (long long)(k0+kk+1)*p.ldw + n0 + nq);
        }
      }
      #pragma unroll
      for(int i=0;i<2;i++){
        int kk = (t>>4)*2 + i*16;
        int nq = (t&15)*4;
        const float* r0 = (const float*)&w0[i];
        const float* r1 = (const float*)&w1[i];
        #pragma unroll
        for(int j=0;j<4;j++){
          bf16x2 o; o[0]=(bf16_t)r0[j]; o[1]=(bf16_t)r1[j];
          *(bf16x2*)(lb + (nq+j)*32 + kk) = o;
        }
      }
    } else {
      // W[k][n] = Wp[n*ldw + k]: contiguous along k -> contiguous lb stores
      float4 wv[4];
      #pragma unroll
      for(int i=0;i<4;i++){
        int n  = i*16 + (t>>3);
        int kq = (t&7)*4;
        if(NB && (n0+n) >= p.N){ wv[i].x=0.f; wv[i].y=0.f; wv[i].z=0.f; wv[i].w=0.f; }
        else wv[i] = *(const float4*)(Wp + (long long)(n0+n)*p.ldw + k0 + kq);
      }
      #pragma unroll
      for(int i=0;i<4;i++){
        int n  = i*16 + (t>>3);
        int kq = (t&7)*4;
        bf16x4 o; o[0]=(bf16_t)wv[i].x; o[1]=(bf16_t)wv[i].y;
                  o[2]=(bf16_t)wv[i].z; o[3]=(bf16_t)wv[i].w;
        *(bf16x4*)(lb + n*32 + kq) = o;
      }
    }
    __syncthreads();

    // ---- fragments (ISA 7.12.2): A lanes0-15 K{0..7,16..23}, lanes16-31 K{8..15,24..31}
    const int mrow = wave*16 + (lane & 15);
    const int kh   = (lane < 16) ? 0 : 8;
    bf16x8 a0 = *(const bf16x8*)(la + mrow*32 + kh);
    bf16x8 a1 = *(const bf16x8*)(la + mrow*32 + 16 + kh);
    v16bf af = __builtin_shufflevector(a0, a1, 0,1,2,3,4,5,6,7,8,9,10,11,12,13,14,15);

    const int ncol = lane & 15;
    const int kh2  = (lane < 16) ? 0 : 8;
    #pragma unroll
    for(int c=0;c<4;c++){
      const bf16_t* bp = lb + (c*16 + ncol)*32 + kh2;
      bf16x8 b0 = *(const bf16x8*)bp;
      bf16x8 b1 = *(const bf16x8*)(bp + 16);
      v16bf bfr = __builtin_shufflevector(b0, b1, 0,1,2,3,4,5,6,7,8,9,10,11,12,13,14,15);
      acc[c] = __builtin_amdgcn_wmma_f32_16x16x32_bf16(
                 false, af, false, bfr, (short)0, acc[c], false, false);
    }
    __syncthreads();
  }

  // ---- epilogue: C/D lane L -> col = L&15, vgpr r -> row = r + 8*(L>>4) ----
  float* Cp = p.C + zq*p.cS1 + zr*p.cS2;
  const float* Rp = p.R ? (p.R + zq*p.rS1 + zr*p.rS2) : nullptr;
  const int mbase = m0 + wave*16 + ((lane>>4)<<3);
  #pragma unroll
  for(int c=0;c<4;c++){
    int n = n0 + c*16 + (lane&15);
    if(NB && n >= p.N) continue;
    #pragma unroll
    for(int r=0;r<8;r++){
      int m = mbase + r;
      float v = p.alpha * acc[c][r];
      if(p.bias) v += p.bias[n];
      if(p.gelu){
        float x = v;
        v = 0.5f*x*(1.f + tanhf(0.7978845608028654f*(x + 0.044715f*x*x*x)));
      }
      if(Rp) v += Rp[(long long)m*p.ldc + n];
      Cp[(long long)m*p.ldc + n] = v;
    }
  }
}

// =====================================================================
// LayerNorm: one wave per 128-wide row; grid.y = batch slice
// =====================================================================
__global__ __launch_bounds__(256) void ln_kernel(
    const float* __restrict__ x, long long xS1,
    float* __restrict__ y, long long yS1,
    const float* __restrict__ s, const float* __restrict__ b, int rows){
  int row  = blockIdx.x*8 + (threadIdx.x>>5);
  int lane = threadIdx.x & 31;
  if(row >= rows) return;
  long long z = blockIdx.y;
  const float* xr = x + z*xS1 + (long long)row*DD + lane*4;
  float4 v = *(const float4*)xr;
  float m = wsum(v.x+v.y+v.z+v.w) * (1.f/128.f);
  float dx=v.x-m, dy=v.y-m, dz=v.z-m, dw=v.w-m;
  float var = wsum(dx*dx+dy*dy+dz*dz+dw*dw) * (1.f/128.f);
  float rs = rsqrtf(var + 1e-5f);
  float4 sv = *(const float4*)(s + lane*4);
  float4 bv = *(const float4*)(b + lane*4);
  float4 o = { dx*rs*sv.x + bv.x, dy*rs*sv.y + bv.y,
               dz*rs*sv.z + bv.z, dw*rs*sv.w + bv.w };
  *(float4*)(y + z*yS1 + (long long)row*DD + lane*4) = o;
}

// =====================================================================
// Fused GATv2: dst = repeat(arange(G),DEG) -> contiguous runs of DEG edges
// One wave per (b,node).  h <- h + gat + bias
// =====================================================================
__global__ __launch_bounds__(256) void gat_kernel(
    float* __restrict__ h, const float* __restrict__ xl,
    const float* __restrict__ xr, const float* __restrict__ att,
    const float* __restrict__ bias, const int* __restrict__ src){
  long long gw = (long long)blockIdx.x*8 + (threadIdx.x>>5);
  int lane = threadIdx.x & 31;
  if(gw >= (long long)BB*GG) return;
  int b = (int)(gw / GG), d = (int)(gw % GG);
  const float* xlb = xl + (long long)b*GG*DD;
  float4 xr4 = *(const float4*)(xr + ((long long)b*GG + d)*DD + lane*4);
  float4 at4 = *(const float4*)(att + lane*4);
  float ej[DEG];
  float mx = -INFINITY;
  for(int j=0;j<DEG;j++){
    int sN = src[(long long)d*DEG + j];
    float4 l4 = *(const float4*)(xlb + (long long)sN*DD + lane*4);
    float ux=l4.x+xr4.x, uy=l4.y+xr4.y, uz=l4.z+xr4.z, uw=l4.w+xr4.w;
    ux = ux>0.f?ux:0.2f*ux; uy = uy>0.f?uy:0.2f*uy;
    uz = uz>0.f?uz:0.2f*uz; uw = uw>0.f?uw:0.2f*uw;
    float e = wsum(ux*at4.x + uy*at4.y + uz*at4.z + uw*at4.w);
    ej[j] = e; mx = fmaxf(mx, e);
  }
  float den = 0.f;
  for(int j=0;j<DEG;j++){ ej[j] = expf(ej[j]-mx); den += ej[j]; }
  float inv = 1.f/(den + 1e-16f);
  float ax=0,ay=0,az=0,aw=0;
  for(int j=0;j<DEG;j++){
    int sN = src[(long long)d*DEG + j];
    float4 l4 = *(const float4*)(xlb + (long long)sN*DD + lane*4);
    float a = ej[j]*inv;
    ax += a*l4.x; ay += a*l4.y; az += a*l4.z; aw += a*l4.w;
  }
  float4 b4 = *(const float4*)(bias + lane*4);
  float* hp = h + ((long long)b*GG + d)*DD + lane*4;
  float4 hc = *(float4*)hp;
  float4 o = { hc.x+ax+b4.x, hc.y+ay+b4.y, hc.z+az+b4.z, hc.w+aw+b4.w };
  *(float4*)hp = o;
}

// =====================================================================
// scores + sort + permute
// =====================================================================
__global__ __launch_bounds__(256) void score_kernel(
    const float* __restrict__ h, const float* __restrict__ sw,
    const float* __restrict__ sb, float* __restrict__ scores){
  long long wv = (long long)blockIdx.x*8 + (threadIdx.x>>5);
  int lane = threadIdx.x & 31;
  if(wv >= (long long)BB*GG) return;
  float4 a = *(const float4*)(h + wv*DD + lane*4);
  float4 w = *(const float4*)(sw + lane*4);
  float s = wsum(a.x*w.x + a.y*w.y + a.z*w.z + a.w*w.w);
  if(lane==0) scores[wv] = s + sb[0];
}

__global__ __launch_bounds__(256) void sortinit_kernel(
    const float* __restrict__ scores, float* __restrict__ skey, int* __restrict__ sidx){
  long long i = (long long)blockIdx.x*blockDim.x + threadIdx.x;
  if(i >= (long long)BB*SORTN) return;
  int b = (int)(i / SORTN), pos = (int)(i % SORTN);
  skey[i] = (pos < GG) ? scores[(long long)b*GG + pos] : -INFINITY;
  sidx[i] = pos;
}

// one block per batch; 32768 keys+idx in 256KB dynamic LDS (<=320KB WGP LDS)
extern __shared__ unsigned char smem_sort[];
__global__ __launch_bounds__(1024) void bitonic_kernel(
    float* __restrict__ skey, int* __restrict__ sidx){
  float* k  = (float*)smem_sort;
  int*   id = (int*)(smem_sort + SORTN*sizeof(float));
  int b = blockIdx.x;
  for(int i=threadIdx.x;i<SORTN;i+=blockDim.x){
    k[i]  = skey[(long long)b*SORTN + i];
    id[i] = sidx[(long long)b*SORTN + i];
  }
  __syncthreads();
  for(int size=2; size<=SORTN; size<<=1){
    for(int stride=size>>1; stride>0; stride>>=1){
      for(int t2=threadIdx.x; t2<SORTN/2; t2+=blockDim.x){
        int i = 2*stride*(t2/stride) + (t2%stride);
        int j = i + stride;
        bool up = (i & size) != 0;          // overall descending
        float ki = k[i], kj = k[j];
        if((ki < kj) != up){
          k[i]=kj; k[j]=ki;
          int tmp=id[i]; id[i]=id[j]; id[j]=tmp;
        }
      }
      __syncthreads();
    }
  }
  for(int i=threadIdx.x;i<SORTN;i+=blockDim.x)
    sidx[(long long)b*SORTN + i] = id[i];
}

__global__ __launch_bounds__(256) void gather_kernel(
    const float* __restrict__ h, const int* __restrict__ sidx, float* __restrict__ xs){
  long long i = (long long)blockIdx.x*blockDim.x + threadIdx.x;
  if(i >= (long long)BB*GG*32) return;
  int c = (int)(i & 31);
  long long t = i >> 5;
  int pos = (int)(t % GG), b = (int)(t / GG);
  int ord = sidx[(long long)b*SORTN + pos];
  ((float4*)xs)[((long long)b*GG + pos)*32 + c] =
      ((const float4*)h)[((long long)b*GG + ord)*32 + c];
}

__global__ __launch_bounds__(256) void scatter_kernel(
    const float* __restrict__ xs, const int* __restrict__ sidx, float* __restrict__ out){
  long long i = (long long)blockIdx.x*blockDim.x + threadIdx.x;
  if(i >= (long long)BB*GG*32) return;
  int c = (int)(i & 31);
  long long t = i >> 5;
  int pos = (int)(t % GG), b = (int)(t / GG);
  int ord = sidx[(long long)b*SORTN + pos];
  ((float4*)out)[((long long)b*GG + ord)*32 + c] =
      ((const float4*)xs)[((long long)b*GG + pos)*32 + c];
}

// =====================================================================
// Performer feature pieces
// =====================================================================
__global__ void feat_init(float* ksum, unsigned* kmax, int nz){
  int i = blockIdx.x*blockDim.x + threadIdx.x;
  if(i < nz*MPAD) ksum[i] = 0.f;
  if(i < nz)      kmax[i] = 0x007FFFFFu;   // fenc(-inf)
}

__global__ __launch_bounds__(256) void kmax_kernel(
    const float* __restrict__ u, unsigned* __restrict__ kmax, int n, int H){
  int z = blockIdx.y;
  long long base = (long long)(z/H)*((long long)H*n*MPAD)
                 + (long long)(z%H)*((long long)n*MPAD);
  long long tot = (long long)n*MPAD;
  float loc = -INFINITY;
  for(long long i=(long long)blockIdx.x*blockDim.x+threadIdx.x; i<tot;
      i += (long long)gridDim.x*blockDim.x){
    int m = (int)(i & (MPAD-1));
    if(m < MFEAT) loc = fmaxf(loc, u[base+i]);
  }
  loc = wmaxr(loc);
  __shared__ float red[8];
  if((threadIdx.x&31)==0) red[threadIdx.x>>5] = loc;
  __syncthreads();
  if(threadIdx.x==0){
    float v = red[0];
    for(int i=1;i<8;i++) v = fmaxf(v, red[i]);
    atomicMax(&kmax[z], fenc(v));
  }
}

// u (in-place -> features). kmax==null: per-row max (query path).
__global__ __launch_bounds__(256) void featexp_kernel(
    float* __restrict__ u, const float* __restrict__ qk,
    const unsigned* __restrict__ kmax, int n, int H){
  int z = blockIdx.y;
  int wv = blockIdx.x*8 + (threadIdx.x>>5);
  int lane = threadIdx.x & 31;
  if(wv >= n) return;
  long long ub = (long long)(z/H)*((long long)H*n*MPAD)
               + (long long)(z%H)*((long long)n*MPAD) + (long long)wv*MPAD;
  const float* qr = qk + (long long)(z/H)*((long long)n*INNER)
                  + (long long)(z%H)*DH + (long long)wv*INNER;
  float qv = qr[lane] * 0.42044820762685725f;    // DH^-0.25
  float diag = 0.5f * wsum(qv*qv);
  float4 uv = *(const float4*)(u + ub + lane*4);
  int mb = lane*4;
  float mx;
  if(kmax){
    mx = fdec(kmax[z]);
  } else {
    float a = (mb+0<MFEAT)?uv.x:-INFINITY;
    float b = (mb+1<MFEAT)?uv.y:-INFINITY;
    float c = (mb+2<MFEAT)?uv.z:-INFINITY;
    float d = (mb+3<MFEAT)?uv.w:-INFINITY;
    mx = wmaxr(fmaxf(fmaxf(a,b),fmaxf(c,d)));
  }
  const float sc = 0.09534625892455922f;         // MFEAT^-0.5
  float4 o;
  o.x = (mb+0<MFEAT)? (expf(uv.x - diag - mx)+1e-4f)*sc : 0.f;
  o.y = (mb+1<MFEAT)? (expf(uv.y - diag - mx)+1e-4f)*sc : 0.f;
  o.z = (mb+2<MFEAT)? (expf(uv.z - diag - mx)+1e-4f)*sc : 0.f;
  o.w = (mb+3<MFEAT)? (expf(uv.w - diag - mx)+1e-4f)*sc : 0.f;
  *(float4*)(u + ub + lane*4) = o;
}

__global__ __launch_bounds__(128) void ksum_kernel(
    const float* __restrict__ kp, float* __restrict__ ksum, int n, int H){
  int z = blockIdx.y;
  long long base = (long long)(z/H)*((long long)H*n*MPAD)
                 + (long long)(z%H)*((long long)n*MPAD);
  int r0 = blockIdx.x*256, t = threadIdx.x;
  int rend = r0+256 < n ? r0+256 : n;
  float acc = 0.f;
  for(int r=r0;r<rend;r++) acc += kp[base + (long long)r*MPAD + t];
  atomicAdd(&ksum[z*MPAD + t], acc);
}

__global__ __launch_bounds__(256) void dendiv_kernel(
    const float* __restrict__ qp, const float* __restrict__ ksum,
    float* __restrict__ num, int n, int H){
  int z = blockIdx.y;
  int wv = blockIdx.x*8 + (threadIdx.x>>5);
  int lane = threadIdx.x & 31;
  if(wv >= n) return;
  const float* qr = qp + (long long)(z/H)*((long long)H*n*MPAD)
                  + (long long)(z%H)*((long long)n*MPAD) + (long long)wv*MPAD;
  const float* ks = ksum + z*MPAD;
  float4 a = *(const float4*)(qr + lane*4);
  float4 w = *(const float4*)(ks + lane*4);
  float den = wsum(a.x*w.x + a.y*w.y + a.z*w.z + a.w*w.w);
  float* nr = num + (long long)(z/H)*((long long)n*INNER)
            + (long long)(z%H)*DH + (long long)wv*INNER;
  nr[lane] = nr[lane] / den;
}

// =====================================================================
// Host-side orchestration
// =====================================================================
static void launch_gemm(hipStream_t st,
    const float* A,int lda,long long aS1,long long aS2,int tA,
    const float* W,int ldw,long long wS1,long long wS2,int tW,
    float* C,int ldc,long long cS1,long long cS2,
    const float* R,long long rS1,long long rS2,
    const float* bias,int M,int N,int K,int nz,int zsub,int gelu,float alpha){
  GemmP p{A,lda,aS1,aS2, W,ldw,wS1,wS2, C,ldc,cS1,cS2,
          R,rS1,rS2, bias, M,N,K,zsub,gelu,alpha};
  dim3 grid(CDIV(M,64), CDIV(N,64), nz);
  dim3 blk(128);
  bool nb = (N % 64) != 0;
  if(tA)            gemm_bf16_wmma<true ,false,true ><<<grid, blk, 0, st>>>(p);
  else if(tW)       gemm_bf16_wmma<false,true ,true ><<<grid, blk, 0, st>>>(p);
  else if(nb)       gemm_bf16_wmma<false,false,true ><<<grid, blk, 0, st>>>(p);
  else              gemm_bf16_wmma<false,false,false><<<grid, blk, 0, st>>>(p);
}

// param order inside each perf dict:
// 0 ln1_s 1 ln1_b 2 wq 3 wk 4 wv 5 wo 6 bo 7 ln2_s 8 ln2_b 9 w1 10 b1 11 w2 12 b2
static void performer(hipStream_t st, float* X, long long xS1, int n, int H,
                      const float* const* P, const float* proj,
                      float* xln, float* q, float* k, float* v, float* num,
                      float* qp, float* kp, float* hid,
                      float* ctx, float* ksum, unsigned* kmax,
                      float* Cfin, long long cfS1){
  const int nz  = BB;
  const int nzh = BB*H;
  const long long snI = (long long)n*INNER;
  const long long snM = (long long)n*MPAD;
  const float fscale = 0.42044820762685725f;   // DH^-0.25 folded into u-GEMM

  // LN1
  ln_kernel<<<dim3(CDIV(n,8),BB), 256, 0, st>>>(X, xS1, xln, (long long)n*DD, P[0], P[1], n);
  // q,k,v projections (K=128 -> N=64)
  launch_gemm(st, xln,DD,(long long)n*DD,0,0, P[2],INNER,0,0,0, q,INNER,snI,0,
              nullptr,0,0, nullptr, n,INNER,DD, nz,1,0, 1.f);
  launch_gemm(st, xln,DD,(long long)n*DD,0,0, P[3],INNER,0,0,0, k,INNER,snI,0,
              nullptr,0,0, nullptr, n,INNER,DD, nz,1,0, 1.f);
  launch_gemm(st, xln,DD,(long long)n*DD,0,0, P[4],INNER,0,0,0, v,INNER,snI,0,
              nullptr,0,0, nullptr, n,INNER,DD, nz,1,0, 1.f);
  // u = (t * DH^-0.25) @ proj^T   (K=32, one WMMA deep)
  launch_gemm(st, q,INNER,snI,DH,0, proj,DH,0,0,1, qp,MPAD,(long long)H*snM,snM,
              nullptr,0,0, nullptr, n,MFEAT,DH, nzh,H,0, fscale);
  launch_gemm(st, k,INNER,snI,DH,0, proj,DH,0,0,1, kp,MPAD,(long long)H*snM,snM,
              nullptr,0,0, nullptr, n,MFEAT,DH, nzh,H,0, fscale);
  // query features: per-row max
  featexp_kernel<<<dim3(CDIV(n,8),nzh), 256, 0, st>>>(qp, q, nullptr, n, H);
  // key features: global (n x m) max per (b,h)
  feat_init<<<CDIV(nzh*MPAD,256), 256, 0, st>>>(ksum, kmax, nzh);
  kmax_kernel<<<dim3(32,nzh), 256, 0, st>>>(kp, kmax, n, H);
  featexp_kernel<<<dim3(CDIV(n,8),nzh), 256, 0, st>>>(kp, k, kmax, n, H);
  // ksum (column sums of kp)
  ksum_kernel<<<dim3(CDIV(n,256),nzh), 128, 0, st>>>(kp, ksum, n, H);
  // ctx = kp^T @ v   (M=128pad, N=32, K=n)
  launch_gemm(st, kp,MPAD,(long long)H*snM,snM,1, v,INNER,snI,DH,0,
              ctx,DH,(long long)H*MPAD*DH,(long long)MPAD*DH,
              nullptr,0,0, nullptr, MPAD,DH,n, nzh,H,0, 1.f);
  // num = qp @ ctx  (into (B,n,H*32) layout)
  launch_gemm(st, qp,MPAD,(long long)H*snM,snM,0, ctx,DH,(long long)H*MPAD*DH,(long long)MPAD*DH,0,
              num,INNER,snI,DH, nullptr,0,0, nullptr, n,DH,MPAD, nzh,H,0, 1.f);
  // den + divide
  dendiv_kernel<<<dim3(CDIV(n,8),nzh), 256, 0, st>>>(qp, ksum, num, n, H);
  // X <- X + o @ wo + bo
  launch_gemm(st, num,INNER,snI,0,0, P[5],DD,0,0,0, X,DD,xS1,0,
              X,xS1,0, P[6], n,DD,INNER, nz,1,0, 1.f);
  // LN2 + FFN
  ln_kernel<<<dim3(CDIV(n,8),BB), 256, 0, st>>>(X, xS1, xln, (long long)n*DD, P[7], P[8], n);
  launch_gemm(st, xln,DD,0,0,0, P[9],FFNH,0,0,0, hid,FFNH,0,0,
              nullptr,0,0, P[10], BB*n,FFNH,DD, 1,1,1, 1.f);          // gelu
  launch_gemm(st, hid,FFNH,(long long)n*FFNH,0,0, P[11],DD,0,0,0, Cfin,DD,cfS1,0,
              X,xS1,0, P[12], n,DD,FFNH, nz,1,0, 1.f);
}

// input index map (setup_inputs insertion-order flatten)
#define IN_X 0
#define IN_LN_S 1
#define IN_LN_B 2
#define IN_GAT_WL 3
#define IN_GAT_WR 4
#define IN_GAT_ATT 5
#define IN_GAT_BIAS 6
#define IN_SCORE_W 7
#define IN_SCORE_B 8
#define IN_BPERF0 9      /* 13 entries per perf dict */
#define IN_FPERF 87
#define IN_PROJB0 100
#define IN_PROJF 106
#define IN_EDGE 107

extern "C" void kernel_launch(void* const* d_in, const int* in_sizes, int n_in,
                              void* d_out, int out_size, void* d_ws, size_t ws_size,
                              hipStream_t stream){
  (void)in_sizes; (void)n_in; (void)out_size; (void)ws_size;
  const float* x    = (const float*)d_in[IN_X];
  const int*   src  = (const int*)d_in[IN_EDGE];          // edge_index[0] = src
  float*       out  = (float*)d_out;
  float*       ws   = (float*)d_ws;

  const long long F = (long long)BB*GG*DD;   // 9,437,184 floats
  float* h    = ws;                 // LN(x) -> +gat -> scatter target -> full-layer state
  float* xl   = ws + F;             // gat xl   (later: xln)
  float* xr   = ws + 2*F;           // gat xr   (later: q @ +0, k @ +F/2)
  float* xs   = ws + 3*F;           // sorted activations (bin layers in place)
  float* vnum = ws + 4*F;           // v @ +0, num @ +F/2
  float* big  = ws + 5*F;           // qp @ +0, kp @ +2F  |  ffn hidden (4F) aliased
  float* misc = ws + 9*F;
  float* scores = misc;
  float* skey   = misc + 131072;
  int*   sidx   = (int*)(misc + 262144);
  float* ctx    = misc + 393216;            // B*H*128*32
  float* ksum   = misc + 393216 + 32768;
  unsigned* kmax = (unsigned*)(ksum + 2048);

  float* xln = xl;
  float* q   = xr;
  float* k   = xr + F/2;
  float* v   = vnum;
  float* num = vnum + F/2;
  float* qp  = big;
  float* kp  = big + 2*F;
  float* hid = big;                 // reused after qp/kp are dead

  // ---- front end: LN -> GAT -> score -> sort -> gather ----
  ln_kernel<<<dim3(CDIV(GG,8),BB), 256, 0, stream>>>(
      x, (long long)GG*DD, h, (long long)GG*DD,
      (const float*)d_in[IN_LN_S], (const float*)d_in[IN_LN_B], GG);
  launch_gemm(stream, h,DD,0,0,0, (const float*)d_in[IN_GAT_WL],DD,0,0,0,
              xl,DD,0,0, nullptr,0,0, nullptr, BB*GG,DD,DD, 1,1,0, 1.f);
  launch_gemm(stream, h,DD,0,0,0, (const float*)d_in[IN_GAT_WR],DD,0,0,0,
              xr,DD,0,0, nullptr,0,0, nullptr, BB*GG,DD,DD, 1,1,0, 1.f);
  gat_kernel<<<CDIV(BB*GG,8), 256, 0, stream>>>(
      h, xl, xr, (const float*)d_in[IN_GAT_ATT], (const float*)d_in[IN_GAT_BIAS], src);
  score_kernel<<<CDIV(BB*GG,8), 256, 0, stream>>>(
      h, (const float*)d_in[IN_SCORE_W], (const float*)d_in[IN_SCORE_B], scores);
  sortinit_kernel<<<CDIV(BB*SORTN,256), 256, 0, stream>>>(scores, skey, sidx);
  bitonic_kernel<<<BB, 1024, SORTN*8, stream>>>(skey, sidx);
  gather_kernel<<<CDIV(BB*GG*32,256), 256, 0, stream>>>(h, sidx, xs);

  // ---- binned performer layers (in place on xs) ----
  for(int i=0;i<NBINS;i++){
    const float* P[13];
    for(int j=0;j<13;j++) P[j] = (const float*)d_in[IN_BPERF0 + 13*i + j];
    const float* proj = (const float*)d_in[IN_PROJB0 + i];
    float* Xb = xs + (long long)i*NBIN*DD;
    performer(stream, Xb, (long long)GG*DD, NBIN, HEADS, P, proj,
              xln, q, k, v, num, qp, kp, hid, ctx, ksum, kmax,
              Xb, (long long)GG*DD);
  }

  // ---- scatter back to natural order (into h), full performer -> d_out ----
  scatter_kernel<<<CDIV(BB*GG*32,256), 256, 0, stream>>>(xs, sidx, h);
  {
    const float* P[13];
    for(int j=0;j<13;j++) P[j] = (const float*)d_in[IN_FPERF + j];
    const float* proj = (const float*)d_in[IN_PROJF];
    performer(stream, h, (long long)GG*DD, GG, HEADS, P, proj,
              xln, q, k, v, num, qp, kp, hid, ctx, ksum, kmax,
              out, (long long)GG*DD);
  }
}
